// GraphW_p_net_24756191494491
// MI455X (gfx1250) — matrix operand order repaired and checked by reading, not compiled, not run on hardware
//
#include <hip/hip_runtime.h>

// ---------------- constants ----------------
#define Bv 32
#define Nn 300
#define Tt 12
#define INDIM 11
#define Cc 32
#define SKIPC 256
#define ENDC 512
#define OUTDIM 3
#define KPAD 320   // node dim padded to 32 (used for both K and N tiling: 20 N-tiles)
#define NTILES 40  // 2 M-tiles x 20 N-tiles
#define NT_PER_WAVE 5

typedef _Float16 h16 __attribute__((ext_vector_type(16)));
typedef _Float16 h8  __attribute__((ext_vector_type(8)));
typedef float    f8  __attribute__((ext_vector_type(8)));

// ---------------- WMMA helpers (CDNA5 16x16x32 f16 -> f32) ----------------
__device__ inline f8 wmma_f32_f16(h16 a, h16 b, f8 c) {
  return __builtin_amdgcn_wmma_f32_16x16x32_f16(
      /*neg_a=*/false, a, /*neg_b=*/false, b,
      /*c_mod=*/(short)0, c, /*reuse_a=*/false, /*reuse_b=*/false);
}

// A fragment: 16x32 f16 tile from row-major src (rows = M, cols = K).
// Lanes 0-15: K=kb+0..15 contiguous; lanes 16-31: K=kb+16..31.
__device__ inline h16 load_a_frag(const _Float16* base, int row_stride,
                                  int m0, int kb, int lane) {
  int half = lane >> 4, mr = lane & 15;
  const _Float16* r = base + (size_t)(m0 + mr) * row_stride;
  h8 lo = *(const h8*)(r + kb + half * 16);
  h8 hi = *(const h8*)(r + kb + half * 16 + 8);
  return __builtin_shufflevector(lo, hi, 0,1,2,3,4,5,6,7,8,9,10,11,12,13,14,15);
}

// B fragment: 32x16 f16 tile, operand stored TRANSPOSED (bt[n][k] row-major).
// Lane holds column n = n0 + lane%16, K range kb + (lane/16)*16 .. +15.
__device__ inline h16 load_b_frag(const _Float16* bt, int row_stride,
                                  int n0, int kb, int lane) {
  int half = lane >> 4, nc = lane & 15;
  return *(const h16*)(bt + (size_t)(n0 + nc) * row_stride + kb + half * 16);
}

// Layer-0 phase-masked B fragment from transposed adjacency aT[w][v].
// keep element iff (adj_phase[v][w]==active[b,t,v] && edge) || v==w
__device__ inline h16 load_b_frag_masked(const _Float16* bt,
                                         const unsigned char* phT,
                                         const unsigned char* actrow,
                                         int n0, int kb, int lane) {
  int half = lane >> 4, nc = lane & 15;
  int n = n0 + nc;
  int k0 = kb + half * 16;
  h16 v = *(const h16*)(bt + (size_t)n * KPAD + k0);
  const unsigned char* ph = phT + (size_t)n * KPAD + k0;
  const unsigned char* ac = actrow + k0;
#pragma unroll
  for (int j = 0; j < 16; ++j) {
    int vv = k0 + j;
    bool keep = (ph[j] == ac[j]) || (vv == n);
    if (!keep) v[j] = (_Float16)0;
  }
  return v;
}

// Store D fragment to row-major f16 and/or transposed f16 LDS buffers.
__device__ inline void store_d(f8 acc, _Float16* rowm, int rm_stride,
                               _Float16* trans, int tr_stride,
                               int m0, int n0, int lane) {
  int half = lane >> 4, nc = lane & 15;
  int n = n0 + nc;
#pragma unroll
  for (int r = 0; r < 8; ++r) {
    int m = m0 + r + half * 8;
    _Float16 hv = (_Float16)acc[r];
    if (rowm)  rowm[(size_t)m * rm_stride + n] = hv;
    if (trans) trans[(size_t)n * tr_stride + m] = hv;
  }
}

// ---------------- prep kernels ----------------
// supports: a0=asym_adj(A), a1=asym_adj(A^T), a2=softmax(relu(nv1@nv2),axis=1)
// stored TRANSPOSED + padded: aT[s][w][v], f16, [3][KPAD][KPAD]
__global__ __launch_bounds__(512) void k_supports(const float* __restrict__ A,
                                                  const float* __restrict__ nv1,
                                                  const float* __restrict__ nv2,
                                                  _Float16* __restrict__ aT) {
  int tid = threadIdx.x;
  for (int idx = tid; idx < 3 * KPAD * KPAD; idx += 512) aT[idx] = (_Float16)0;
  __syncthreads();
  if (tid >= Nn) return;
  int v = tid;
  float rs = 0.f, cs = 0.f;
  for (int w = 0; w < Nn; ++w) { rs += A[v * Nn + w]; cs += A[w * Nn + v]; }
  float dr = rs > 0.f ? 1.f / rs : 0.f;
  float dc = cs > 0.f ? 1.f / cs : 0.f;
  for (int w = 0; w < Nn; ++w) {
    aT[0 * KPAD * KPAD + w * KPAD + v] = (_Float16)(A[v * Nn + w] * dr);
    aT[1 * KPAD * KPAD + w * KPAD + v] = (_Float16)(A[w * Nn + v] * dc);
  }
  // adaptive support row v: softmax over w of relu(nv1[v]@nv2[:,w])
  float mx = 0.f;  // relu >= 0 so max >= 0
  for (int w = 0; w < Nn; ++w) {
    float z = 0.f;
    for (int k = 0; k < 10; ++k) z += nv1[v * 10 + k] * nv2[k * Nn + w];
    z = z > 0.f ? z : 0.f;
    mx = z > mx ? z : mx;
  }
  float se = 0.f;
  for (int w = 0; w < Nn; ++w) {
    float z = 0.f;
    for (int k = 0; k < 10; ++k) z += nv1[v * 10 + k] * nv2[k * Nn + w];
    z = z > 0.f ? z : 0.f;
    se += expf(z - mx);
  }
  float inv = 1.f / se;
  for (int w = 0; w < Nn; ++w) {
    float z = 0.f;
    for (int k = 0; k < 10; ++k) z += nv1[v * 10 + k] * nv2[k * Nn + w];
    z = z > 0.f ? z : 0.f;
    aT[2 * KPAD * KPAD + w * KPAD + v] = (_Float16)(expf(z - mx) * inv);
  }
}

// phT[w][v] = phase if edge v->w exists else 255 (never matches active 0..7)
__global__ void k_prep_ph(const float* __restrict__ adj,
                          const int* __restrict__ phase,
                          unsigned char* __restrict__ phT) {
  int idx = blockIdx.x * 256 + threadIdx.x;
  if (idx >= KPAD * KPAD) return;
  int v = idx % KPAD, w = idx / KPAD;
  unsigned char out = 255;
  if (v < Nn && w < Nn && adj[v * Nn + w] > 0.f)
    out = (unsigned char)phase[v * Nn + w];
  phT[idx] = out;
}

// act[b][t][v] = argmax_k input[b, 3+k, v, t]   (255 in pad region)
__global__ void k_prep_act(const float* __restrict__ input,
                           unsigned char* __restrict__ act) {
  int idx = blockIdx.x * 256 + threadIdx.x;
  if (idx >= Bv * Tt * KPAD) return;
  int v = idx % KPAD, t = (idx / KPAD) % Tt, b = idx / (KPAD * Tt);
  unsigned char out = 255;
  if (v < Nn) {
    int best = 0;
    float bv = input[((b * INDIM + 3) * Nn + v) * Tt + t];
    for (int k = 1; k < 8; ++k) {
      float val = input[((b * INDIM + 3 + k) * Nn + v) * Tt + t];
      if (val > bv) { bv = val; best = k; }
    }
    out = (unsigned char)best;
  }
  act[idx] = out;
}

__global__ void k_prep_w16(const float* __restrict__ w, _Float16* __restrict__ w16) {
  int idx = blockIdx.x * 256 + threadIdx.x;
  if (idx < 8 * Cc * 7 * Cc) w16[idx] = (_Float16)w[idx];
}

// start conv on left-padded input: x0[b][c][n][t], t=0..12, t==0 is zero pad
__global__ void k_start(const float* __restrict__ input,
                        const float* __restrict__ W,
                        const float* __restrict__ bias,
                        float* __restrict__ x0) {
  int idx = blockIdx.x * 256 + threadIdx.x;
  const int L = 13;
  if (idx >= Bv * Cc * Nn * L) return;
  int t = idx % L, n = (idx / L) % Nn, c = (idx / (L * Nn)) % Cc, b = idx / (L * Nn * Cc);
  float acc = bias[c];
  if (t >= 1) {
    for (int ic = 0; ic < INDIM; ++ic)
      acc += W[c * INDIM + ic] * input[((b * INDIM + ic) * Nn + n) * Tt + (t - 1)];
  }
  x0[idx] = acc;
}

// ---------------- per-layer kernels ----------------
// gated TCN: xg = tanh(filt(x)) * sigmoid(gate(x)), kernel (1,2) dilation d
__global__ void k_tcn(const float* __restrict__ xin,
                      const float* __restrict__ fW, const float* __restrict__ fb,
                      const float* __restrict__ gW, const float* __restrict__ gb,
                      float* __restrict__ gated,
                      int layer, int Lin, int Lout, int d) {
  int idx = blockIdx.x * 256 + threadIdx.x;
  if (idx >= Bv * Cc * Nn * Lout) return;
  int l = idx % Lout, n = (idx / Lout) % Nn, co = (idx / (Lout * Nn)) % Cc,
      b = idx / (Lout * Nn * Cc);
  float sf = fb[layer * Cc + co];
  float sg = gb[layer * Cc + co];
  const float* fw = fW + ((size_t)layer * Cc + co) * Cc * 2;
  const float* gw = gW + ((size_t)layer * Cc + co) * Cc * 2;
  for (int ci = 0; ci < Cc; ++ci) {
    float x0 = xin[((b * Cc + ci) * Nn + n) * Lin + l];
    float x1 = xin[((b * Cc + ci) * Nn + n) * Lin + l + d];
    sf += fw[ci * 2 + 0] * x0 + fw[ci * 2 + 1] * x1;
    sg += gw[ci * 2 + 0] * x0 + gw[ci * 2 + 1] * x1;
  }
  gated[idx] = tanhf(sf) * (1.f / (1.f + expf(-sg)));
}

// skip accumulator: only the LAST time step of each layer survives
__global__ void k_skip(const float* __restrict__ gated, float* __restrict__ skip,
                       const float* __restrict__ sW, const float* __restrict__ sb,
                       int layer, int Lout, int init) {
  int idx = blockIdx.x * 256 + threadIdx.x;
  if (idx >= Bv * SKIPC * Nn) return;
  int n = idx % Nn, sc = (idx / Nn) % SKIPC, b = idx / (Nn * SKIPC);
  float acc = sb[layer * SKIPC + sc];
  const float* w = sW + ((size_t)layer * SKIPC + sc) * Cc;
  for (int c = 0; c < Cc; ++c)
    acc += w[c] * gated[((b * Cc + c) * Nn + n) * Lout + (Lout - 1)];
  if (init) skip[idx] = acc;
  else      skip[idx] += acc;
}

// Fused graph-conv: for one (b,l) slice compute x1,x2 for 3 supports with
// WMMA over the node dimension, accumulate the 224->32 gconv in D registers,
// add residual, write layer output. MASKED=true applies the layer-0 per-(b,t)
// phase mask while building B fragments.
// Fixed 40 tiles (2 Mtiles x 20 Ntiles) = exactly 5 per wave -> accumulators
// stay in VGPRs (no dynamic indexing, no scratch spill).
template <bool MASKED>
__global__ __launch_bounds__(256) void k_gconv(
    const float* __restrict__ gated, const float* __restrict__ resid,
    float* __restrict__ xout, const _Float16* __restrict__ aT,
    const unsigned char* __restrict__ phT, const unsigned char* __restrict__ act,
    const _Float16* __restrict__ Wg16, const float* __restrict__ gcb,
    int layer, int Lin, int Lout) {
  extern __shared__ _Float16 sm[];
  _Float16* xs  = sm;                  // [Cc][KPAD]   gated slice, A layout
  _Float16* xst = xs  + Cc * KPAD;     // [KPAD][Cc]   gated slice, B layout
  _Float16* h1  = xst + KPAD * Cc;     // [Cc][KPAD]   hop1, A layout
  _Float16* h1t = h1  + Cc * KPAD;     // [KPAD][Cc]   hop1, B layout
  _Float16* h2t = h1t + KPAD * Cc;     // [KPAD][Cc]   hop2, B layout

  int b = blockIdx.x / Lout, l = blockIdx.x % Lout;
  int tid = threadIdx.x, lane = tid & 31, wave = tid >> 5;

  for (int idx = tid; idx < Cc * KPAD; idx += 256) {
    int c = idx / KPAD, k = idx % KPAD;
    float v = (k < Nn) ? gated[((b * Cc + c) * Nn + k) * Lout + l] : 0.f;
    xs[idx] = (_Float16)v;
    xst[k * Cc + c] = (_Float16)v;
  }
  __syncthreads();

  const unsigned char* actrow = act + ((size_t)b * Tt + l) * KPAD;  // layer 0 only
  const _Float16* Wrow = Wg16 + (size_t)layer * Cc * (7 * Cc);

  // tile t = wave + 8*j : m0 = (t&1)*16, n0 = (t>>1)*16
  int tm[NT_PER_WAVE], tn[NT_PER_WAVE];
#pragma unroll
  for (int j = 0; j < NT_PER_WAVE; ++j) {
    int t = wave + 8 * j;
    tm[j] = (t & 1) * 16;
    tn[j] = (t >> 1) * 16;
  }

  // chunk 0: out = Wg[:,0:32] @ x
  f8 o0, o1, o2, o3, o4;
  {
    f8 z = {0.f, 0.f, 0.f, 0.f, 0.f, 0.f, 0.f, 0.f};
    o0 = wmma_f32_f16(load_a_frag(Wrow, 7 * Cc, tm[0], 0, lane),
                      load_b_frag(xst, Cc, tn[0], 0, lane), z);
    o1 = wmma_f32_f16(load_a_frag(Wrow, 7 * Cc, tm[1], 0, lane),
                      load_b_frag(xst, Cc, tn[1], 0, lane), z);
    o2 = wmma_f32_f16(load_a_frag(Wrow, 7 * Cc, tm[2], 0, lane),
                      load_b_frag(xst, Cc, tn[2], 0, lane), z);
    o3 = wmma_f32_f16(load_a_frag(Wrow, 7 * Cc, tm[3], 0, lane),
                      load_b_frag(xst, Cc, tn[3], 0, lane), z);
    o4 = wmma_f32_f16(load_a_frag(Wrow, 7 * Cc, tm[4], 0, lane),
                      load_b_frag(xst, Cc, tn[4], 0, lane), z);
  }

  for (int s = 0; s < 3; ++s) {
    const _Float16* aTs = aT + (size_t)s * KPAD * KPAD;
    __syncthreads();
    // P1: x1 = x * a  -> h1 (A layout), h1t (B layout)
#pragma unroll
    for (int j = 0; j < NT_PER_WAVE; ++j) {
      int m0 = tm[j], n0 = tn[j];
      f8 acc = {0.f, 0.f, 0.f, 0.f, 0.f, 0.f, 0.f, 0.f};
      for (int kb = 0; kb < KPAD; kb += 32) {
        h16 a = load_a_frag(xs, KPAD, m0, kb, lane);
        h16 bf = MASKED ? load_b_frag_masked(aTs, phT, actrow, n0, kb, lane)
                        : load_b_frag(aTs, KPAD, n0, kb, lane);
        acc = wmma_f32_f16(a, bf, acc);
      }
      store_d(acc, h1, KPAD, h1t, Cc, m0, n0, lane);
    }
    __syncthreads();
    // P2: out += Wg[:,(1+2s)*32 ...] @ x1 ; x2 = x1 * a -> h2t
    {
      int c0 = (1 + 2 * s) * Cc;
      o0 = wmma_f32_f16(load_a_frag(Wrow, 7 * Cc, tm[0], c0, lane),
                        load_b_frag(h1t, Cc, tn[0], 0, lane), o0);
      o1 = wmma_f32_f16(load_a_frag(Wrow, 7 * Cc, tm[1], c0, lane),
                        load_b_frag(h1t, Cc, tn[1], 0, lane), o1);
      o2 = wmma_f32_f16(load_a_frag(Wrow, 7 * Cc, tm[2], c0, lane),
                        load_b_frag(h1t, Cc, tn[2], 0, lane), o2);
      o3 = wmma_f32_f16(load_a_frag(Wrow, 7 * Cc, tm[3], c0, lane),
                        load_b_frag(h1t, Cc, tn[3], 0, lane), o3);
      o4 = wmma_f32_f16(load_a_frag(Wrow, 7 * Cc, tm[4], c0, lane),
                        load_b_frag(h1t, Cc, tn[4], 0, lane), o4);
#pragma unroll
      for (int j = 0; j < NT_PER_WAVE; ++j) {
        int m0 = tm[j], n0 = tn[j];
        f8 acc = {0.f, 0.f, 0.f, 0.f, 0.f, 0.f, 0.f, 0.f};
        for (int kb = 0; kb < KPAD; kb += 32) {
          h16 a = load_a_frag(h1, KPAD, m0, kb, lane);
          h16 bf = MASKED ? load_b_frag_masked(aTs, phT, actrow, n0, kb, lane)
                          : load_b_frag(aTs, KPAD, n0, kb, lane);
          acc = wmma_f32_f16(a, bf, acc);
        }
        store_d(acc, nullptr, 0, h2t, Cc, m0, n0, lane);
      }
    }
    __syncthreads();
    // P3: out += Wg[:,(2+2s)*32 ...] @ x2
    {
      int c0 = (2 + 2 * s) * Cc;
      o0 = wmma_f32_f16(load_a_frag(Wrow, 7 * Cc, tm[0], c0, lane),
                        load_b_frag(h2t, Cc, tn[0], 0, lane), o0);
      o1 = wmma_f32_f16(load_a_frag(Wrow, 7 * Cc, tm[1], c0, lane),
                        load_b_frag(h2t, Cc, tn[1], 0, lane), o1);
      o2 = wmma_f32_f16(load_a_frag(Wrow, 7 * Cc, tm[2], c0, lane),
                        load_b_frag(h2t, Cc, tn[2], 0, lane), o2);
      o3 = wmma_f32_f16(load_a_frag(Wrow, 7 * Cc, tm[3], c0, lane),
                        load_b_frag(h2t, Cc, tn[3], 0, lane), o3);
      o4 = wmma_f32_f16(load_a_frag(Wrow, 7 * Cc, tm[4], c0, lane),
                        load_b_frag(h2t, Cc, tn[4], 0, lane), o4);
    }
  }

  // bias + residual (trim: resid[..., Lin-Lout+l]) + store (skip pad tiles)
  int dL = Lin - Lout;
  f8 oacc[NT_PER_WAVE] = {o0, o1, o2, o3, o4};
#pragma unroll
  for (int j = 0; j < NT_PER_WAVE; ++j) {
    int m0 = tm[j], n0 = tn[j];
    int half = lane >> 4, nc = lane & 15;
    int n = n0 + nc;
    if (n < Nn) {
#pragma unroll
      for (int r = 0; r < 8; ++r) {
        int m = m0 + r + half * 8;
        float v = oacc[j][r] + gcb[layer * Cc + m] +
                  resid[((b * Cc + m) * Nn + n) * Lin + (l + dL)];
        xout[((b * Cc + m) * Nn + n) * Lout + l] = v;
      }
    }
  }
}

// training-mode BN stats per channel over (B,N,L)
__global__ __launch_bounds__(256) void k_bn_stats(const float* __restrict__ x,
                                                  float* __restrict__ stats, int Lout) {
  __shared__ float ssum[256], ssq[256];
  int c = blockIdx.x, tid = threadIdx.x;
  int total = Bv * Nn * Lout;
  float s = 0.f, q = 0.f;
  for (int idx = tid; idx < total; idx += 256) {
    int l = idx % Lout, n = (idx / Lout) % Nn, b = idx / (Lout * Nn);
    float v = x[((b * Cc + c) * Nn + n) * Lout + l];
    s += v; q += v * v;
  }
  ssum[tid] = s; ssq[tid] = q;
  __syncthreads();
  for (int o = 128; o > 0; o >>= 1) {
    if (tid < o) { ssum[tid] += ssum[tid + o]; ssq[tid] += ssq[tid + o]; }
    __syncthreads();
  }
  if (tid == 0) {
    float mu = ssum[0] / total;
    stats[c] = mu;
    stats[Cc + c] = ssq[0] / total - mu * mu;
  }
}

__global__ void k_bn_apply(float* __restrict__ x, const float* __restrict__ stats,
                           const float* __restrict__ g, const float* __restrict__ bb,
                           int Lout) {
  int idx = blockIdx.x * 256 + threadIdx.x;
  if (idx >= Bv * Cc * Nn * Lout) return;
  int c = (idx / (Nn * Lout)) % Cc;
  float mu = stats[c], var = stats[Cc + c];
  float xn = (x[idx] - mu) * rsqrtf(var + 1e-5f);
  x[idx] = xn * g[c] + bb[c];
}

// head: relu(skip) -> end1(512) relu -> end2(3)
__global__ __launch_bounds__(512) void k_head(const float* __restrict__ skip,
                                              const float* __restrict__ W1,
                                              const float* __restrict__ b1,
                                              const float* __restrict__ W2,
                                              const float* __restrict__ b2,
                                              float* __restrict__ out) {
  __shared__ float h[ENDC];
  int bn = blockIdx.x;
  int b = bn / Nn, n = bn % Nn;
  int e = threadIdx.x;
  float acc = b1[e];
  const float* w = W1 + (size_t)e * SKIPC;
  for (int k = 0; k < SKIPC; ++k) {
    float sv = skip[(b * SKIPC + k) * Nn + n];
    sv = sv > 0.f ? sv : 0.f;
    acc += w[k] * sv;
  }
  h[e] = acc > 0.f ? acc : 0.f;
  __syncthreads();
  if (e < OUTDIM) {
    float o = b2[e];
    const float* w2 = W2 + (size_t)e * ENDC;
    for (int k = 0; k < ENDC; ++k) o += w2[k] * h[k];
    out[(b * OUTDIM + e) * Nn + n] = o;
  }
}

// ---------------- launch ----------------
extern "C" void kernel_launch(void* const* d_in, const int* in_sizes, int n_in,
                              void* d_out, int out_size, void* d_ws, size_t ws_size,
                              hipStream_t stream) {
  const float* input  = (const float*)d_in[0];
  const float* adj    = (const float*)d_in[1];
  const int*   phase  = (const int*)d_in[2];
  const float* startW = (const float*)d_in[3];
  const float* startb = (const float*)d_in[4];
  const float* filtW  = (const float*)d_in[5];
  const float* filtb  = (const float*)d_in[6];
  const float* gateW  = (const float*)d_in[7];
  const float* gateb  = (const float*)d_in[8];
  const float* skipW  = (const float*)d_in[9];
  const float* skipb  = (const float*)d_in[10];
  const float* gconvW = (const float*)d_in[11];
  const float* gconvb = (const float*)d_in[12];
  const float* bng    = (const float*)d_in[13];
  const float* bnb    = (const float*)d_in[14];
  const float* nv1    = (const float*)d_in[15];
  const float* nv2    = (const float*)d_in[16];
  const float* e1W    = (const float*)d_in[17];
  const float* e1b    = (const float*)d_in[18];
  const float* e2W    = (const float*)d_in[19];
  const float* e2b    = (const float*)d_in[20];

  char* ws = (char*)d_ws;
  size_t off = 0;
  auto take = [&](size_t bytes) -> void* {
    void* p = ws + off;
    off = (off + bytes + 255) & ~(size_t)255;
    return p;
  };
  float*         xA    = (float*)take((size_t)Bv * Cc * Nn * 13 * sizeof(float));
  float*         xB    = (float*)take((size_t)Bv * Cc * Nn * 13 * sizeof(float));
  float*         gated = (float*)take((size_t)Bv * Cc * Nn * 12 * sizeof(float));
  float*         skp   = (float*)take((size_t)Bv * SKIPC * Nn * sizeof(float));
  _Float16*      aT    = (_Float16*)take((size_t)3 * KPAD * KPAD * sizeof(_Float16));
  unsigned char* phT   = (unsigned char*)take((size_t)KPAD * KPAD);
  unsigned char* act   = (unsigned char*)take((size_t)Bv * Tt * KPAD);
  _Float16*      Wg16  = (_Float16*)take((size_t)8 * Cc * 7 * Cc * sizeof(_Float16));
  float*         stats = (float*)take(64 * sizeof(float));

  k_supports<<<1, 512, 0, stream>>>(adj, nv1, nv2, aT);
  k_prep_ph<<<(KPAD * KPAD + 255) / 256, 256, 0, stream>>>(adj, phase, phT);
  k_prep_act<<<(Bv * Tt * KPAD + 255) / 256, 256, 0, stream>>>(input, act);
  k_prep_w16<<<(8 * Cc * 7 * Cc + 255) / 256, 256, 0, stream>>>(gconvW, Wg16);
  k_start<<<(Bv * Cc * Nn * 13 + 255) / 256, 256, 0, stream>>>(input, startW, startb, xA);

  const int dil[8] = {1, 2, 1, 2, 1, 2, 1, 2};
  float* xin = xA;
  float* xout = xB;
  int L = 13;
  size_t shbytes = (size_t)(5 * Cc * KPAD) * sizeof(_Float16);  // 100 KB

  for (int i = 0; i < 8; ++i) {
    int d = dil[i];
    int Lout = L - d;
    int ntcn = Bv * Cc * Nn * Lout;
    k_tcn<<<(ntcn + 255) / 256, 256, 0, stream>>>(xin, filtW, filtb, gateW, gateb,
                                                  gated, i, L, Lout, d);
    k_skip<<<(Bv * SKIPC * Nn + 255) / 256, 256, 0, stream>>>(gated, skp, skipW, skipb,
                                                              i, Lout, i == 0 ? 1 : 0);
    if (i == 0) {
      k_gconv<true><<<Bv * Lout, 256, shbytes, stream>>>(gated, xin, xout, aT, phT, act,
                                                         Wg16, gconvb, i, L, Lout);
    } else {
      k_gconv<false><<<Bv * Lout, 256, shbytes, stream>>>(gated, xin, xout, aT, phT, act,
                                                          Wg16, gconvb, i, L, Lout);
    }
    k_bn_stats<<<Cc, 256, 0, stream>>>(xout, stats, Lout);
    k_bn_apply<<<(ntcn + 255) / 256, 256, 0, stream>>>(xout, stats, bng + i * Cc,
                                                       bnb + i * Cc, Lout);
    float* t = xin; xin = xout; xout = t;
    L = Lout;
  }

  k_head<<<Bv * Nn, 512, 0, stream>>>(skp, e1W, e1b, e2W, e2b, (float*)d_out);
}
